// TopKRouter_67851893342554
// MI455X (gfx1250) — compile-verified, matrix operand order checked by reading, not run on hardware
//
#include <hip/hip_runtime.h>
#include <hip/hip_bf16.h>

typedef __attribute__((ext_vector_type(16))) __bf16 v16bf;
typedef __attribute__((ext_vector_type(8)))  float  v8f;

#define BLOCK_M   128          // tokens per workgroup (8 waves x 16)
#define WAVE_M    16
#define NUM_E     64
#define LDS_STRIDE 68          // 128x68 f32 tile; row-half offset 8*68 % 64 = 32 banks -> no conflicts

// Build a 16x32 bf16 WMMA A/B fragment from a row-major f32 row.
// ISA 7.12.2 (16-bit 16x32): lane holds row M = lane%16; lanes 0-15 carry
// K in [0,8)+[16,24), lanes 16-31 carry K in [8,16)+[24,32).
// Caller passes k = k0 + (lane>>4)*8; elements 8..15 live at k+16.
__device__ __forceinline__ v16bf load_frag_bf16(const float* __restrict__ base, int k) {
    const float4 p0 = *(const float4*)(base + k);
    const float4 p1 = *(const float4*)(base + k + 4);
    const float4 p2 = *(const float4*)(base + k + 16);
    const float4 p3 = *(const float4*)(base + k + 20);
    v16bf v;
    v[0]  = (__bf16)p0.x; v[1]  = (__bf16)p0.y; v[2]  = (__bf16)p0.z; v[3]  = (__bf16)p0.w;
    v[4]  = (__bf16)p1.x; v[5]  = (__bf16)p1.y; v[6]  = (__bf16)p1.z; v[7]  = (__bf16)p1.w;
    v[8]  = (__bf16)p2.x; v[9]  = (__bf16)p2.y; v[10] = (__bf16)p2.z; v[11] = (__bf16)p2.w;
    v[12] = (__bf16)p3.x; v[13] = (__bf16)p3.y; v[14] = (__bf16)p3.z; v[15] = (__bf16)p3.w;
    return v;
}

__global__ __launch_bounds__(256, 2)
void topk_router_kernel(const float* __restrict__ x,
                        const float* __restrict__ W,
                        const float* __restrict__ bias,
                        float* __restrict__ mixing,
                        float* __restrict__ probs,
                        float* __restrict__ idx_out,   // indices as f32 (concat promotion)
                        float* __restrict__ wt_out,
                        int N, int D)
{
    __shared__ float sLog[BLOCK_M][LDS_STRIDE];
    __shared__ float rMax[BLOCK_M], rInv[BLOCK_M], rW1[BLOCK_M], rW2[BLOCK_M];
    __shared__ int   rI1[BLOCK_M], rI2[BLOCK_M];

    const int tid    = threadIdx.x;
    const int wave   = tid >> 5;
    const int lane   = tid & 31;
    const int laneLo = lane & 15;
    const int laneHi = lane >> 4;

    const int rowBase = blockIdx.x * BLOCK_M + wave * WAVE_M;

    const float* xrow = x + (size_t)(rowBase + laneLo) * D;
    const float* wp0  = W + (size_t)(laneLo +  0) * D;
    const float* wp1  = W + (size_t)(laneLo + 16) * D;
    const float* wp2  = W + (size_t)(laneLo + 32) * D;
    const float* wp3  = W + (size_t)(laneLo + 48) * D;

    v8f c0 = {}, c1 = {}, c2 = {}, c3 = {};

    const int koff = laneHi * 8;

    // ---- GEMM: logits[16 x 64] per wave, K-chunks of 32 (bf16 WMMA, f32 acc)
    for (int k0 = 0; k0 < D; k0 += 32) {
        // prefetch the x stream ahead (speculative; OOB prefetch is dropped)
        __builtin_prefetch((const void*)(xrow + k0 + 256), 0, 0);

        const int k = k0 + koff;
        v16bf a  = load_frag_bf16(xrow, k);
        v16bf b0 = load_frag_bf16(wp0,  k);
        v16bf b1 = load_frag_bf16(wp1,  k);
        v16bf b2 = load_frag_bf16(wp2,  k);
        v16bf b3 = load_frag_bf16(wp3,  k);

        c0 = __builtin_amdgcn_wmma_f32_16x16x32_bf16(false, a, false, b0, (short)0, c0, false, false);
        c1 = __builtin_amdgcn_wmma_f32_16x16x32_bf16(false, a, false, b1, (short)0, c1, false, false);
        c2 = __builtin_amdgcn_wmma_f32_16x16x32_bf16(false, a, false, b2, (short)0, c2, false, false);
        c3 = __builtin_amdgcn_wmma_f32_16x16x32_bf16(false, a, false, b3, (short)0, c3, false, false);
    }

    // ---- bias add + spill logits tile to LDS
    // C/D layout: lanes 0-15 hold rows 0..7 in v0..7 (col = lane), lanes 16-31 rows 8..15.
    {
        const float bv0 = bias[laneLo +  0];
        const float bv1 = bias[laneLo + 16];
        const float bv2 = bias[laneLo + 32];
        const float bv3 = bias[laneLo + 48];
        #pragma unroll
        for (int j = 0; j < 8; j++) {
            const int rr = wave * WAVE_M + laneHi * 8 + j;
            sLog[rr][laneLo +  0] = c0[j] + bv0;
            sLog[rr][laneLo + 16] = c1[j] + bv1;
            sLog[rr][laneLo + 32] = c2[j] + bv2;
            sLog[rr][laneLo + 48] = c3[j] + bv3;
        }
    }
    __syncthreads();

    // ---- per-token softmax stats + stable top-2 (ties -> lower index, like lax.top_k)
    if (tid < BLOCK_M) {
        const float* lrow = sLog[tid];
        float m = lrow[0];
        #pragma unroll 8
        for (int e = 1; e < NUM_E; e++) m = fmaxf(m, lrow[e]);

        float sum = 0.0f;
        float v1 = -3.0e38f, v2 = -3.0e38f;
        int   i1 = 0, i2 = 0;
        #pragma unroll 4
        for (int e = 0; e < NUM_E; e++) {
            const float l = lrow[e];
            sum += __expf(l - m);
            if (l > v1)      { v2 = v1; i2 = i1; v1 = l; i1 = e; }
            else if (l > v2) { v2 = l;  i2 = e; }
        }
        const float inv = 1.0f / sum;
        const float p1 = __expf(v1 - m) * inv;
        const float p2 = __expf(v2 - m) * inv;
        const float dn = p1 + p2 + 1e-9f;
        const float w1 = p1 / dn, w2 = p2 / dn;

        rMax[tid] = m;  rInv[tid] = inv;
        rI1[tid]  = i1; rI2[tid]  = i2;
        rW1[tid]  = w1; rW2[tid]  = w2;

        const int g = blockIdx.x * BLOCK_M + tid;
        idx_out[2 * g + 0] = (float)i1;
        idx_out[2 * g + 1] = (float)i2;
        wt_out [2 * g + 0] = w1;
        wt_out [2 * g + 1] = w2;
    }
    __syncthreads();

    // ---- coalesced writes of probs + dense mixing
    const int base = blockIdx.x * BLOCK_M;
    #pragma unroll
    for (int i = 0; i < (BLOCK_M * NUM_E) / 256; i++) {
        const int idx = i * 256 + tid;
        const int row = idx >> 6;       // NUM_E == 64
        const int col = idx & 63;
        const float p = __expf(sLog[row][col] - rMax[row]) * rInv[row];
        const size_t go = (size_t)(base + row) * NUM_E + col;
        probs[go] = p;
        float mv = 0.0f;
        if (col == rI1[row]) mv = rW1[row];
        else if (col == rI2[row]) mv = rW2[row];
        mixing[go] = mv;
    }
}

extern "C" void kernel_launch(void* const* d_in, const int* in_sizes, int n_in,
                              void* d_out, int out_size, void* d_ws, size_t ws_size,
                              hipStream_t stream) {
    const float* x = (const float*)d_in[0];
    const float* W = (const float*)d_in[1];
    const float* b = (const float*)d_in[2];

    const int E = in_sizes[2];           // 64
    const int D = in_sizes[1] / E;       // 4096
    const int N = in_sizes[0] / D;       // 16384

    float* out     = (float*)d_out;
    float* mixing  = out;                                  // N*E
    float* probs   = out + (size_t)N * E;                  // N*E
    float* idx_out = out + 2 * (size_t)N * E;              // N*2 (as f32)
    float* wt_out  = idx_out + (size_t)N * 2;              // N*2

    dim3 grid(N / BLOCK_M);
    dim3 block(256);
    topk_router_kernel<<<grid, block, 0, stream>>>(x, W, b, mixing, probs, idx_out, wt_out, N, D);
}
